// CGCNN_67018669687321
// MI455X (gfx1250) — compile-verified, matrix operand order checked by reading, not run on hardware
//
#include <hip/hip_runtime.h>
#include <hip/hip_bf16.h>

#define N_NODES  50000   // divisible by 16 -> no M-tail anywhere
#define N_EDGES  800000  // divisible by 16
#define X_DIM    92
#define XK_PAD   96
#define EDGE_DIM 80
#define H_DIM    128
#define IN_DIM   336   // 2*H + EDGE_DIM
#define K_PAD    352   // IN_DIM padded to /32
#define KSTEPS   (K_PAD / 32)   // 11
#define QPR      (K_PAD / 4)    // 88 quads per A-tile row
#define TWO_H    256
#define N_GRAPHS 256
#define BN_EPS   1e-5f

typedef __attribute__((ext_vector_type(16))) _Float16 v16h;
typedef __attribute__((ext_vector_type(4)))  _Float16 v4h;
typedef __attribute__((ext_vector_type(8)))  float    v8f;

__device__ __forceinline__ v8f wmma_f32_16x16x32(v16h a, v16h b, v8f c) {
  // D = A(16x32 f16) * B(32x16 f16) + C(16x16 f32)
  return __builtin_amdgcn_wmma_f32_16x16x32_f16(false, a, false, b, (short)0, c,
                                                false, false);
}

// A fragment: 16x32 f16 tile, row-major in LDS with leading dim lda.
// Per ISA 7.12.2 each lane's 16 halves are two contiguous 8-half runs -> 2x ds_load_b128.
__device__ __forceinline__ void load_a_frag(const _Float16* __restrict__ As, int lda,
                                            int lane, int k0, v16h& a) {
  const int hs = lane >> 4;
  const int m  = lane & 15;
  const _Float16* row = As + m * lda + k0 + 8 * hs;
#pragma unroll
  for (int v = 0; v < 8; ++v) {
    const int kb = (v < 4) ? (2 * v) : (16 + 2 * (v - 4));
    a[2 * v]     = row[kb];
    a[2 * v + 1] = row[kb + 1];
  }
}

// B fragment straight from fragment-ordered global W:
// Wsw[((s*ntiles + nt)*32 + lane)*16 + slot]  (32B per lane, aligned) -> global_load_b128 x2
__device__ __forceinline__ v16h load_b_direct(const _Float16* __restrict__ Wsw,
                                              int s, int ntiles, int nt, int lane) {
  return *reinterpret_cast<const v16h*>(Wsw + ((size_t)(s * ntiles + nt) * 32 + lane) * 16);
}

__device__ __forceinline__ v4h f4_to_h4(float4 f) {
  v4h h;
  h[0] = (_Float16)f.x; h[1] = (_Float16)f.y;
  h[2] = (_Float16)f.z; h[3] = (_Float16)f.w;
  return h;
}

__device__ __forceinline__ float sigmoidf_(float x) { return 1.f / (1.f + __expf(-x)); }
__device__ __forceinline__ float softplusf_(float x) {
  return x > 20.f ? x : log1pf(__expf(x));
}

// ---------------- utility kernels ----------------

__global__ void k_zero(float* __restrict__ p, int n) {
  for (int i = blockIdx.x * blockDim.x + threadIdx.x; i < n; i += gridDim.x * blockDim.x)
    p[i] = 0.f;
}

// fp32 row-major [rows][cols] -> f16 WMMA-fragment-ordered [ksteps][ntiles][32][16]
// slot v of lane l, n-tile nt, k-step s holds W[k][n] with
//   k = 32*s + (v&7) + 8*(l>>4) + 16*(v>>3),  n = nt*16 + (l&15)
__global__ void k_swizzle_w(const float* __restrict__ src, _Float16* __restrict__ dst,
                            int rows, int cols, int ksteps, int ntiles) {
  const int total = ksteps * ntiles * 32 * 16;
  for (int i = blockIdx.x * blockDim.x + threadIdx.x; i < total; i += gridDim.x * blockDim.x) {
    const int slot = i & 15;
    const int lane = (i >> 4) & 31;
    const int nt   = (i >> 9) % ntiles;
    const int s    = i / (512 * ntiles);
    const int k = 32 * s + (slot & 7) + 8 * (lane >> 4) + 16 * (slot >> 3);
    const int n = nt * 16 + (lane & 15);
    dst[i] = (_Float16)((k < rows) ? src[k * cols + n] : 0.f);
  }
}

// ---------------- embedding GEMM: h = x @ embed_W + b ----------------

__global__ __launch_bounds__(256) void k_embed(const float* __restrict__ x,
                                               const _Float16* __restrict__ Wsw, // [3][8][32][16]
                                               const float* __restrict__ bias,
                                               float* __restrict__ hOut) {
  __shared__ _Float16 As[16 * XK_PAD];
  const int tid = threadIdx.x;
  const int m0  = blockIdx.x * 16;  // 3125*16 == 50000: never out of range

  // gather A tile [16][96] f16 (X_DIM=92, pad 92..95); 92 = 23 quads exactly
  for (int q = tid; q < 16 * (XK_PAD / 4); q += 256) {
    const int r = q / (XK_PAD / 4), qc = q - r * (XK_PAD / 4);
    const int c = 4 * qc;
    v4h hv = {};
    if (c < X_DIM) {
      // X_DIM=92 not /16: rows are 368B apart, a 16B multiple, so quads keep x's alignment; scalar path
      const float* p = x + (size_t)(m0 + r) * X_DIM + c;
      hv[0] = (_Float16)p[0];
      hv[1] = (_Float16)p[1];
      hv[2] = (_Float16)p[2];
      hv[3] = (_Float16)(c + 3 < X_DIM ? p[3] : 0.f);
    }
    *reinterpret_cast<v4h*>(As + r * XK_PAD + c) = hv;
  }
  __syncthreads();

  const int wave = tid >> 5, lane = tid & 31;
  v8f c = {};
  v16h a, b;
#pragma unroll
  for (int s = 0; s < XK_PAD / 32; ++s) {
    load_a_frag(As, XK_PAD, lane, 32 * s, a);
    b = load_b_direct(Wsw, s, 8, wave, lane);
    c = wmma_f32_16x16x32(a, b, c);
  }
  const int n  = wave * 16 + (lane & 15);
  const int hs = lane >> 4;
  const float bn = bias[n];
#pragma unroll
  for (int r = 0; r < 8; ++r)
    hOut[(size_t)(m0 + r + 8 * hs) * H_DIM + n] = c[r] + bn;
}

// ---------------- per-edge message GEMM fragments (shared by both passes) -------
// c0/c1 hold msg tile columns [32w,32w+16) / [32w+16,32w+32) (pre-bias).
// Gather is 4-wide: concat regions 128|80|128 are all /4, so a quad never
// straddles a region -> one global_load_b128 + one ds_store_b64 per quad.

__device__ __forceinline__ void compute_msg_frags(
    int e0, const float* __restrict__ h, const float* __restrict__ eattr,
    const int* __restrict__ src, const int* __restrict__ dst,
    const _Float16* __restrict__ Wsw, _Float16* As, v8f& c0, v8f& c1) {
  const int tid = threadIdx.x;
  __builtin_prefetch(Wsw + (size_t)tid * 64, 0, 0);  // warm L2/GL1 for W

  for (int q = tid; q < 16 * QPR; q += 256) {
    const int r  = q / QPR, qc = q - r * QPR;
    const int c  = 4 * qc;
    const int e  = e0 + r;
    v4h hv = {};
    if (c < H_DIM) {
      hv = f4_to_h4(*reinterpret_cast<const float4*>(h + (size_t)src[e] * H_DIM + c));
    } else if (c < H_DIM + EDGE_DIM) {
      hv = f4_to_h4(*reinterpret_cast<const float4*>(eattr + (size_t)e * EDGE_DIM + (c - H_DIM)));
    } else if (c < IN_DIM) {
      hv = f4_to_h4(*reinterpret_cast<const float4*>(h + (size_t)dst[e] * H_DIM +
                                                     (c - H_DIM - EDGE_DIM)));
    }
    *reinterpret_cast<v4h*>(As + r * K_PAD + c) = hv;
  }
  __syncthreads();

  const int wave = tid >> 5, lane = tid & 31;
  c0 = (v8f){}; c1 = (v8f){};
  v16h a, b0, b1;
#pragma unroll
  for (int s = 0; s < KSTEPS; ++s) {
    load_a_frag(As, K_PAD, lane, 32 * s, a);
    b0 = load_b_direct(Wsw, s, 16, 2 * wave, lane);
    b1 = load_b_direct(Wsw, s, 16, 2 * wave + 1, lane);
    c0 = wmma_f32_16x16x32(a, b0, c0);
    c1 = wmma_f32_16x16x32(a, b1, c1);
  }
}

// pass 1: accumulate per-column sum / sumsq of msg (BN batch statistics).
// Each lane owns 8 rows of one column; xor-shuffle folds the two row groups.
__global__ __launch_bounds__(256) void k_msg_stats(
    const float* __restrict__ h, const float* __restrict__ eattr,
    const int* __restrict__ edge_index, const _Float16* __restrict__ Wsw,
    const float* __restrict__ bias, float* __restrict__ msum, float* __restrict__ msumsq) {
  __shared__ _Float16 As[16 * K_PAD];
  const int e0 = blockIdx.x * 16;
  v8f c0, c1;
  compute_msg_frags(e0, h, eattr, edge_index, edge_index + N_EDGES, Wsw, As, c0, c1);

  const int tid = threadIdx.x;
  const int wave = tid >> 5, lane = tid & 31;
  const int nA = 32 * wave + (lane & 15);
  const int nB = nA + 16;
  const float bA = bias[nA], bB = bias[nB];
  float s = 0.f, s2 = 0.f, t = 0.f, t2 = 0.f;
#pragma unroll
  for (int r = 0; r < 8; ++r) {
    const float v = c0[r] + bA; s += v; s2 += v * v;
    const float u = c1[r] + bB; t += u; t2 += u * u;
  }
  s  += __shfl_xor(s, 16);  s2 += __shfl_xor(s2, 16);
  t  += __shfl_xor(t, 16);  t2 += __shfl_xor(t2, 16);
  if ((lane >> 4) == 0) {
    atomicAdd(&msum[nA], s);  atomicAdd(&msumsq[nA], s2);
    atomicAdd(&msum[nB], t);  atomicAdd(&msumsq[nB], t2);
  }
}

__global__ void k_bn_finalize(const float* __restrict__ sum, const float* __restrict__ sumsq,
                              float* __restrict__ mean, float* __restrict__ inv,
                              int n, float count) {
  const int j = blockIdx.x * blockDim.x + threadIdx.x;
  if (j < n) {
    const float mu  = sum[j] / count;
    const float var = sumsq[j] / count - mu * mu;
    mean[j] = mu;
    inv[j]  = rsqrtf(var + BN_EPS);
  }
}

// pass 2: recompute msg, apply BN + sigmoid(filt)*softplus(core), scatter into agg[src]
__global__ __launch_bounds__(256) void k_msg_apply(
    const float* __restrict__ h, const float* __restrict__ eattr,
    const int* __restrict__ edge_index, const _Float16* __restrict__ Wsw,
    const float* __restrict__ bias, const float* __restrict__ mean,
    const float* __restrict__ inv, const float* __restrict__ gamma,
    const float* __restrict__ beta, float* __restrict__ agg) {
  __shared__ _Float16 As[16 * K_PAD];
  __shared__ float    Ms[16 * TWO_H];
  const int e0 = blockIdx.x * 16;
  const int* src = edge_index;
  v8f c0, c1;
  compute_msg_frags(e0, h, eattr, src, edge_index + N_EDGES, Wsw, As, c0, c1);

  const int tid = threadIdx.x;
  const int wave = tid >> 5, lane = tid & 31;
  {
    const int n  = lane & 15;
    const int hs = lane >> 4;
    const int nA = 32 * wave + n;
    const int nB = nA + 16;
    const float bA = bias[nA], bB = bias[nB];
#pragma unroll
    for (int r = 0; r < 8; ++r) {
      const int m = r + 8 * hs;
      Ms[m * TWO_H + nA] = c0[r] + bA;
      Ms[m * TWO_H + nB] = c1[r] + bB;
    }
  }
  __syncthreads();

  const int j = tid & 127;
  const float muF = mean[j],       ivF = inv[j],       gF = gamma[j],       bF = beta[j];
  const float muC = mean[j + 128], ivC = inv[j + 128], gC = gamma[j + 128], bC = beta[j + 128];
  for (int e = tid >> 7; e < 16; e += 2) {
    const float f = (Ms[e * TWO_H + j]       - muF) * ivF * gF + bF;
    const float c = (Ms[e * TWO_H + 128 + j] - muC) * ivC * gC + bC;
    const float g = sigmoidf_(f) * softplusf_(c);
    atomicAdd(&agg[(size_t)src[e0 + e] * H_DIM + j], g);
  }
}

// column sum / sumsq over [rows][128]
__global__ __launch_bounds__(256) void k_col_stats128(const float* __restrict__ a, int rows,
                                                      float* __restrict__ sum,
                                                      float* __restrict__ sumsq) {
  __shared__ float s1[H_DIM], s2[H_DIM];
  const int tid = threadIdx.x;
  if (tid < H_DIM) { s1[tid] = 0.f; s2[tid] = 0.f; }
  __syncthreads();
  const int total = rows * H_DIM;
  for (int i = blockIdx.x * blockDim.x + tid; i < total; i += gridDim.x * blockDim.x) {
    const float v = a[i];
    const int c = i & (H_DIM - 1);
    atomicAdd(&s1[c], v);
    atomicAdd(&s2[c], v * v);
  }
  __syncthreads();
  if (tid < H_DIM) { atomicAdd(&sum[tid], s1[tid]); atomicAdd(&sumsq[tid], s2[tid]); }
}

// h_new = softplus(h + bn(agg))
__global__ void k_node_update(const float* __restrict__ hIn, const float* __restrict__ agg,
                              const float* __restrict__ mean, const float* __restrict__ inv,
                              const float* __restrict__ gamma, const float* __restrict__ beta,
                              float* __restrict__ hOut) {
  const int total = N_NODES * H_DIM;
  for (int i = blockIdx.x * blockDim.x + threadIdx.x; i < total; i += gridDim.x * blockDim.x) {
    const int j = i & (H_DIM - 1);
    const float a = (agg[i] - mean[j]) * inv[j] * gamma[j] + beta[j];
    hOut[i] = softplusf_(hIn[i] + a);
  }
}

// mean pooling scatter
__global__ void k_pool(const float* __restrict__ h, const int* __restrict__ batch,
                       float* __restrict__ pooled, float* __restrict__ cnt) {
  const int total = N_NODES * H_DIM;
  for (int i = blockIdx.x * blockDim.x + threadIdx.x; i < total; i += gridDim.x * blockDim.x) {
    const int node = i >> 7, j = i & (H_DIM - 1);
    const int g = batch[node];
    atomicAdd(&pooled[g * H_DIM + j], h[i]);
    if (j == 0) atomicAdd(&cnt[g], 1.f);
  }
}

// head: out[g] = softplus(pooled/cnt @ linW + linb) @ outW + outb
__global__ __launch_bounds__(128) void k_head(const float* __restrict__ pooled,
                                              const float* __restrict__ cnt,
                                              const float* __restrict__ linW,
                                              const float* __restrict__ linb,
                                              const float* __restrict__ outW,
                                              const float* __restrict__ outb,
                                              float* __restrict__ out) {
  __shared__ float p[H_DIM];
  __shared__ float t[H_DIM];
  const int g = blockIdx.x, j = threadIdx.x;
  const float c = fmaxf(cnt[g], 1.f);
  p[j] = pooled[g * H_DIM + j] / c;
  __syncthreads();
  float acc = linb[j];
#pragma unroll 4
  for (int k = 0; k < H_DIM; ++k) acc += p[k] * linW[k * H_DIM + j];
  t[j] = softplusf_(acc) * outW[j];
  __syncthreads();
  for (int off = 64; off > 0; off >>= 1) {
    if (j < off) t[j] += t[j + off];
    __syncthreads();
  }
  if (j == 0) out[g] = t[0] + outb[0];
}

// ---------------- host orchestration ----------------

extern "C" void kernel_launch(void* const* d_in, const int* in_sizes, int n_in,
                              void* d_out, int out_size, void* d_ws, size_t ws_size,
                              hipStream_t stream) {
  (void)in_sizes; (void)n_in; (void)out_size; (void)ws_size;
  const float* x     = (const float*)d_in[0];
  const int*   ei    = (const int*)d_in[1];
  const float* eattr = (const float*)d_in[2];
  const int*   batch = (const int*)d_in[3];
  const float* embW  = (const float*)d_in[4];
  const float* embB  = (const float*)d_in[5];
  const float* convW = (const float*)d_in[6];
  const float* convB = (const float*)d_in[7];
  const float* bnig  = (const float*)d_in[8];
  const float* bnib  = (const float*)d_in[9];
  const float* bnog  = (const float*)d_in[10];
  const float* bnob  = (const float*)d_in[11];
  const float* linW  = (const float*)d_in[12];
  const float* linb  = (const float*)d_in[13];
  const float* outW  = (const float*)d_in[14];
  const float* outb  = (const float*)d_in[15];

  char* ws = (char*)d_ws;
  size_t off = 0;
  auto alloc = [&](size_t bytes) -> void* {
    void* p = ws + off;
    off = (off + bytes + 255) & ~(size_t)255;
    return p;
  };
  float*    hA     = (float*)alloc(sizeof(float) * N_NODES * H_DIM);
  float*    hB     = (float*)alloc(sizeof(float) * N_NODES * H_DIM);
  float*    agg    = (float*)alloc(sizeof(float) * N_NODES * H_DIM);
  _Float16* Wsw    = (_Float16*)alloc(sizeof(_Float16) * KSTEPS * 16 * 32 * 16);
  _Float16* eWsw   = (_Float16*)alloc(sizeof(_Float16) * 3 * 8 * 32 * 16);
  float*    msum   = (float*)alloc(sizeof(float) * TWO_H);
  float*    msumsq = (float*)alloc(sizeof(float) * TWO_H);
  float*    mmean  = (float*)alloc(sizeof(float) * TWO_H);
  float*    minv   = (float*)alloc(sizeof(float) * TWO_H);
  float*    asum   = (float*)alloc(sizeof(float) * H_DIM);
  float*    asumsq = (float*)alloc(sizeof(float) * H_DIM);
  float*    amean  = (float*)alloc(sizeof(float) * H_DIM);
  float*    ainv   = (float*)alloc(sizeof(float) * H_DIM);
  float*    pooled = (float*)alloc(sizeof(float) * N_GRAPHS * H_DIM);
  float*    cnt    = (float*)alloc(sizeof(float) * N_GRAPHS);

  // embedding
  k_swizzle_w<<<48, 256, 0, stream>>>(embW, eWsw, X_DIM, H_DIM, 3, 8);
  k_embed<<<N_NODES / 16, 256, 0, stream>>>(x, eWsw, embB, hA);

  float* bufs[2] = {hA, hB};
  int cur = 0;
  for (int L = 0; L < 3; ++L) {
    const float* hIn  = bufs[cur];
    float*       hOut = bufs[1 - cur];
    const float* W = convW + (size_t)L * IN_DIM * TWO_H;
    const float* b = convB + L * TWO_H;

    k_swizzle_w<<<352, 256, 0, stream>>>(W, Wsw, IN_DIM, TWO_H, KSTEPS, 16);
    k_zero<<<1, 256, 0, stream>>>(msum, TWO_H);
    k_zero<<<1, 256, 0, stream>>>(msumsq, TWO_H);
    k_zero<<<1, 256, 0, stream>>>(asum, H_DIM);
    k_zero<<<1, 256, 0, stream>>>(asumsq, H_DIM);
    k_zero<<<2048, 256, 0, stream>>>(agg, N_NODES * H_DIM);

    k_msg_stats<<<N_EDGES / 16, 256, 0, stream>>>(hIn, eattr, ei, Wsw, b, msum, msumsq);
    k_bn_finalize<<<1, 256, 0, stream>>>(msum, msumsq, mmean, minv, TWO_H, (float)N_EDGES);
    k_msg_apply<<<N_EDGES / 16, 256, 0, stream>>>(hIn, eattr, ei, Wsw, b, mmean, minv,
                                                  bnig + L * TWO_H, bnib + L * TWO_H, agg);
    k_col_stats128<<<1024, 256, 0, stream>>>(agg, N_NODES, asum, asumsq);
    k_bn_finalize<<<1, 128, 0, stream>>>(asum, asumsq, amean, ainv, H_DIM, (float)N_NODES);
    k_node_update<<<2048, 256, 0, stream>>>(hIn, agg, amean, ainv, bnog + L * H_DIM,
                                            bnob + L * H_DIM, hOut);
    cur = 1 - cur;
  }

  const float* hFinal = bufs[cur];
  k_zero<<<128, 256, 0, stream>>>(pooled, N_GRAPHS * H_DIM);
  k_zero<<<1, 256, 0, stream>>>(cnt, N_GRAPHS);
  k_pool<<<2048, 256, 0, stream>>>(hFinal, batch, pooled, cnt);
  k_head<<<N_GRAPHS, 128, 0, stream>>>(pooled, cnt, linW, linb, outW, outb, (float*)d_out);
}
